// CrossAttentionLayer_50517405335834
// MI455X (gfx1250) — matrix operand order repaired and checked by reading, not compiled, not run on hardware
//
#include <hip/hip_runtime.h>
#include <hip/hip_bf16.h>

// ---------------------------------------------------------------------------
// CrossAttentionLayer — single fused kernel for gfx1250 (MI455X), plus a tiny
// weight prep kernel (fp32 -> bf16, transposed to [N][K]) run once per launch.
//
//  Per WG (64 tokens, 256 thr = 8 waves):
//    stage X1/X2 (bf16, LDS) -> WMMA Q/K/V (bf16 out, LDS)
//    -> per-token 4x4 scores+softmax (register resident)
//    -> context (bf16, LDS) -> WMMA out-proj (A straight from LDS)
//    -> LayerNorm -> d_out.   B fragments come directly from L2-resident
//    pre-transposed bf16 weights (two global b128 loads per fragment);
//    A fragments are two ds_load_b128 each.  No syncthreads in GEMM loops.
// ---------------------------------------------------------------------------

#define OCC_DIM   256
#define COORD_DIM 64
#define OUT_DIM   512
#define N_HEADS   4
#define HEAD_DIM  128
#define TOTAL_ROWS (32 * 4096)            // 131072
#define ROWS_PER_WG 64
#define NUM_WGS   (TOTAL_ROWS / ROWS_PER_WG) // 2048

// Transposed bf16 weight offsets inside d_ws (u16 elements)
#define WQT_OFF 0               // [512][256]
#define WKT_OFF 131072          // [512][64]
#define WVT_OFF 163840          // [512][64]
#define WOT_OFF 196608          // [512][512]
#define WT_TOTAL 458752         // u16 elements (917504 bytes)

// LDS map (bytes)
#define S_X1   0                // u16 [64][264]  33792  (later overlaid by sO)
#define S_X2   33792            // u16 [64][72]    9216
#define S_Q    43008            // u16 [64][520]  66560
#define S_K    109568           // u16 [64][520]  66560
#define S_V    176128           // u16 [64][520]  66560
#define S_S    242688           // f32 [64][16]    4096
#define S_CTX  246784           // u16 [64][520]  66560
#define SMEM_SZ 313344          // <= 320 KB
#define S_O    0                // f32 [64][516] 132096 (overlays X1/X2/Q/K)

typedef __attribute__((ext_vector_type(16))) __bf16 v16bf;
typedef __attribute__((ext_vector_type(8)))  float  v8f;

union Frag { v16bf bf; unsigned short u[16]; uint4 q[2]; };

static __device__ __forceinline__ unsigned short f2bf(float f) {
    unsigned int x = __float_as_uint(f);
    unsigned int r = x + 0x7FFFu + ((x >> 16) & 1u);   // RNE
    return (unsigned short)(r >> 16);
}
static __device__ __forceinline__ unsigned int pack2(float a, float b) {
    return (unsigned int)f2bf(a) | ((unsigned int)f2bf(b) << 16);
}
static __device__ __forceinline__ float bflo(unsigned int u) {
    return __uint_as_float(u << 16);
}
static __device__ __forceinline__ float bfhi(unsigned int u) {
    return __uint_as_float(u & 0xFFFF0000u);
}
static __device__ __forceinline__ void unpack8(uint4 v, float* f) {
    f[0] = bflo(v.x); f[1] = bfhi(v.x); f[2] = bflo(v.y); f[3] = bfhi(v.y);
    f[4] = bflo(v.z); f[5] = bfhi(v.z); f[6] = bflo(v.w); f[7] = bfhi(v.w);
}
static __device__ __forceinline__ v8f wmma_bf16(v16bf a, v16bf b, v8f c) {
    return __builtin_amdgcn_wmma_f32_16x16x32_bf16(
        false, a, false, b, (short)0, c, false, false);
}

// 64xKdim (LDS bf16, row-major) @ WT [512][Kdim] (global bf16, pre-transposed)
// -> 64x512 (+bias).  Wave w owns cols [w*64, w*64+64).  No syncthreads.
template<bool OUT_BF16>
static __device__ __forceinline__ void gemm64(
    const unsigned short* __restrict__ sA, int ldA, int Kdim,
    const unsigned short* __restrict__ WT,
    const float* __restrict__ bias,
    void* __restrict__ sOutv, int ldO,
    int wid, int lane)
{
    v8f zero = {0.f,0.f,0.f,0.f,0.f,0.f,0.f,0.f};
    v8f acc[4][4];
#pragma unroll
    for (int mt = 0; mt < 4; ++mt)
#pragma unroll
        for (int nt = 0; nt < 4; ++nt) acc[mt][nt] = zero;

    const int mrow = lane & 15;
    const int kbA  = (lane >> 4) << 3;   // 0 or 8
    const int kbB  = (lane >> 4) << 3;   // 0 or 8
    const int ncol = lane & 15;

    for (int k0 = 0; k0 < Kdim; k0 += 32) {
        Frag a[4];
#pragma unroll
        for (int mt = 0; mt < 4; ++mt) {
            const unsigned short* p = sA + (mt * 16 + mrow) * ldA + k0 + kbA;
            a[mt].q[0] = *(const uint4*)p;          // K: kbA .. kbA+7
            a[mt].q[1] = *(const uint4*)(p + 16);   // K: kbA+16 .. kbA+23
        }
#pragma unroll
        for (int nt = 0; nt < 4; ++nt) {
            int n = wid * 64 + nt * 16 + ncol;
            const unsigned short* p = WT + (size_t)n * Kdim + k0 + kbB;
            Frag b;
            b.q[0] = *(const uint4*)p;              // K: kbB .. kbB+7
            b.q[1] = *(const uint4*)(p + 16);       // K: kbB+16 .. kbB+23
#pragma unroll
            for (int mt = 0; mt < 4; ++mt)
                acc[mt][nt] = wmma_bf16(a[mt].bf, b.bf, acc[mt][nt]);
        }
    }

    const int mtop = (lane >> 4) << 3;   // 0 or 8
#pragma unroll
    for (int nt = 0; nt < 4; ++nt) {
        int n = wid * 64 + nt * 16 + ncol;
        float bv = bias[n];
#pragma unroll
        for (int mt = 0; mt < 4; ++mt)
#pragma unroll
            for (int r = 0; r < 8; ++r) {
                float v = acc[mt][nt][r] + bv;
                int m = mt * 16 + mtop + r;
                if constexpr (OUT_BF16)
                    ((unsigned short*)sOutv)[m * ldO + n] = f2bf(v);
                else
                    ((float*)sOutv)[m * ldO + n] = v;
            }
    }
}

// --------------------------- weight prep kernel ----------------------------
__global__ __launch_bounds__(256)
void prep_weights_kernel(const float* __restrict__ Wq, const float* __restrict__ Wk,
                         const float* __restrict__ Wv, const float* __restrict__ Wo,
                         unsigned short* __restrict__ wt)
{
    int idx = blockIdx.x * 256 + threadIdx.x;          // 0 .. 458751
    if (idx < 131072) {                                // Wq [256][512] -> [512][256]
        int k = idx >> 9, n = idx & 511;
        wt[WQT_OFF + n * 256 + k] = f2bf(Wq[idx]);
    } else if (idx < 163840) {                         // Wk [64][512] -> [512][64]
        int j = idx - 131072; int k = j >> 9, n = j & 511;
        wt[WKT_OFF + n * 64 + k] = f2bf(Wk[j]);
    } else if (idx < 196608) {                         // Wv
        int j = idx - 163840; int k = j >> 9, n = j & 511;
        wt[WVT_OFF + n * 64 + k] = f2bf(Wv[j]);
    } else if (idx < WT_TOTAL) {                       // Wo [512][512] -> [512][512]
        int j = idx - 196608; int k = j >> 9, n = j & 511;
        wt[WOT_OFF + n * 512 + k] = f2bf(Wo[j]);
    }
}

// ------------------------------ fused kernel -------------------------------
__global__ __launch_bounds__(256)
void ca_fused_kernel(const float* __restrict__ occ, const float* __restrict__ coord,
                     const unsigned short* __restrict__ wt,
                     const float* __restrict__ bq, const float* __restrict__ bk,
                     const float* __restrict__ bv, const float* __restrict__ bo,
                     const float* __restrict__ gamma, const float* __restrict__ beta,
                     float* __restrict__ out, float* __restrict__ attn_out)
{
    extern __shared__ unsigned char smem[];
    unsigned short* sX1 = (unsigned short*)(smem + S_X1);
    unsigned short* sX2 = (unsigned short*)(smem + S_X2);
    unsigned short* sQ  = (unsigned short*)(smem + S_Q);
    unsigned short* sK  = (unsigned short*)(smem + S_K);
    unsigned short* sV  = (unsigned short*)(smem + S_V);
    float*          sS  = (float*)(smem + S_S);
    unsigned short* sC  = (unsigned short*)(smem + S_CTX);
    float*          sO  = (float*)(smem + S_O);

    const int tid  = threadIdx.x;
    const int wid  = tid >> 5;
    const int lane = tid & 31;
    const size_t row0 = (size_t)blockIdx.x * ROWS_PER_WG;

    // ---- stage inputs as bf16 (paired u32 stores) ----
    for (int i = tid; i < 64 * OCC_DIM / 2; i += 256) {     // 8192 pairs
        int r = i >> 7, c = (i & 127) * 2;
        const float* p = occ + (row0 + r) * OCC_DIM + c;
        *(unsigned int*)&sX1[r * 264 + c] = pack2(p[0], p[1]);
    }
    for (int i = tid; i < 64 * COORD_DIM / 2; i += 256) {   // 2048 pairs
        int r = i >> 5, c = (i & 31) * 2;
        const float* p = coord + (row0 + r) * COORD_DIM + c;
        *(unsigned int*)&sX2[r * 72 + c] = pack2(p[0], p[1]);
    }
    __syncthreads();

    // ---- Q/K/V projections (bf16 WMMA) ----
    gemm64<true>(sX1, 264, OCC_DIM,   wt + WQT_OFF, bq, sQ, 520, wid, lane);
    gemm64<true>(sX2,  72, COORD_DIM, wt + WKT_OFF, bk, sK, 520, wid, lane);
    gemm64<true>(sX2,  72, COORD_DIM, wt + WVT_OFF, bv, sV, 520, wid, lane);
    __syncthreads();

    // ---- scores + softmax, fully per-thread: (token t, head h) ----
    {
        int t = tid >> 2, h = tid & 3;
        const unsigned short* qrow = sQ + t * 520 + h * HEAD_DIM;
        float dot[4] = {0.f, 0.f, 0.f, 0.f};
#pragma unroll 4
        for (int i = 0; i < 16; ++i) {
            float qf[8];
            unpack8(*(const uint4*)(qrow + i * 8), qf);
#pragma unroll
            for (int g = 0; g < 4; ++g) {
                float kf[8];
                unpack8(*(const uint4*)(sK + t * 520 + g * HEAD_DIM + i * 8), kf);
                float d = 0.f;
#pragma unroll
                for (int j = 0; j < 8; ++j) d += qf[j] * kf[j];
                dot[g] += d;
            }
        }
        const float scale = 0.08838834764831845f;           // 1/sqrt(128)
        float m = fmaxf(fmaxf(dot[0], dot[1]), fmaxf(dot[2], dot[3])) * scale;
        float e0 = __expf(dot[0] * scale - m), e1 = __expf(dot[1] * scale - m);
        float e2 = __expf(dot[2] * scale - m), e3 = __expf(dot[3] * scale - m);
        float inv = 1.0f / (e0 + e1 + e2 + e3);
        sS[t * 16 + h * 4 + 0] = e0 * inv;
        sS[t * 16 + h * 4 + 1] = e1 * inv;
        sS[t * 16 + h * 4 + 2] = e2 * inv;
        sS[t * 16 + h * 4 + 3] = e3 * inv;
    }
    __syncthreads();

    // ---- attn.mean over heads -> second output ----
    {
        int t = tid >> 2, g = tid & 3;
        float m = 0.25f * (sS[t * 16 + 0 + g] + sS[t * 16 + 4 + g] +
                           sS[t * 16 + 8 + g] + sS[t * 16 + 12 + g]);
        attn_out[(row0 + t) * N_HEADS + g] = m;
    }

    // ---- context = attn @ V  (bf16 -> LDS, 8 outputs per step) ----
    for (int it = 0; it < 16; ++it) {
        int cid = it * 256 + tid;          // 4096 chunks of 8
        int t = cid >> 6, cs = (cid & 63) * 8;
        int h = cs >> 7, d0 = cs & 127;
        const float* a = sS + t * 16 + h * 4;
        float o[8] = {0.f,0.f,0.f,0.f,0.f,0.f,0.f,0.f};
#pragma unroll
        for (int g = 0; g < 4; ++g) {
            float vf[8];
            unpack8(*(const uint4*)(sV + t * 520 + g * HEAD_DIM + d0), vf);
            float ag = a[g];
#pragma unroll
            for (int j = 0; j < 8; ++j) o[j] += ag * vf[j];
        }
        uint4 pk;
        pk.x = pack2(o[0], o[1]); pk.y = pack2(o[2], o[3]);
        pk.z = pack2(o[4], o[5]); pk.w = pack2(o[6], o[7]);
        *(uint4*)(sC + t * 520 + cs) = pk;
    }
    __syncthreads();

    // ---- output projection: sO = ctx @ Wo + bo (fp32 epilogue) ----
    gemm64<false>(sC, 520, OUT_DIM, wt + WOT_OFF, bo, sO, 516, wid, lane);
    __syncthreads();

    // ---- LayerNorm: 4 threads per row, 128 contiguous cols each ----
    {
        int t = tid >> 2, sub = tid & 3;
        const float* rowp = sO + t * 516 + sub * 128;
        float s = 0.f, ss = 0.f;
#pragma unroll 8
        for (int j = 0; j < 32; ++j) {
            float4 x = *(const float4*)(rowp + 4 * j);
            s  += x.x + x.y + x.z + x.w;
            ss += x.x * x.x + x.y * x.y + x.z * x.z + x.w * x.w;
        }
        for (int m = 2; m >= 1; m >>= 1) {
            s  += __shfl_xor(s,  m, 4);
            ss += __shfl_xor(ss, m, 4);
        }
        float mu   = s * (1.0f / OUT_DIM);
        float var  = ss * (1.0f / OUT_DIM) - mu * mu;
        float rinv = rsqrtf(var + 1e-5f);
        float* op = out + (row0 + t) * OUT_DIM + sub * 128;
        const float* gp = gamma + sub * 128;
        const float* bp = beta + sub * 128;
#pragma unroll 8
        for (int j = 0; j < 32; ++j) {
            float4 x = *(const float4*)(rowp + 4 * j);
            float4 g = *(const float4*)(gp + 4 * j);
            float4 b = *(const float4*)(bp + 4 * j);
            float4 r;
            r.x = (x.x - mu) * rinv * g.x + b.x;
            r.y = (x.y - mu) * rinv * g.y + b.y;
            r.z = (x.z - mu) * rinv * g.z + b.z;
            r.w = (x.w - mu) * rinv * g.w + b.w;
            *(float4*)(op + 4 * j) = r;
        }
    }
}

// ---------------------------------------------------------------------------
extern "C" void kernel_launch(void* const* d_in, const int* in_sizes, int n_in,
                              void* d_out, int out_size, void* d_ws, size_t ws_size,
                              hipStream_t stream) {
    const float* occ   = (const float*)d_in[0];
    const float* coord = (const float*)d_in[1];
    const float* Wq    = (const float*)d_in[2];
    const float* bq    = (const float*)d_in[3];
    const float* Wk    = (const float*)d_in[4];
    const float* bk    = (const float*)d_in[5];
    const float* Wv    = (const float*)d_in[6];
    const float* bv    = (const float*)d_in[7];
    const float* Wo    = (const float*)d_in[8];
    const float* bo    = (const float*)d_in[9];
    const float* gamma = (const float*)d_in[10];
    const float* beta  = (const float*)d_in[11];

    float* out      = (float*)d_out;
    float* attn_out = out + (size_t)TOTAL_ROWS * OUT_DIM;
    unsigned short* wt = (unsigned short*)d_ws;   // 917504 B of bf16 weights

    (void)hipFuncSetAttribute((const void*)ca_fused_kernel,
                              hipFuncAttributeMaxDynamicSharedMemorySize, SMEM_SZ);

    prep_weights_kernel<<<dim3(WT_TOTAL / 256), dim3(256), 0, stream>>>(
        Wq, Wk, Wv, Wo, wt);
    ca_fused_kernel<<<dim3(NUM_WGS), dim3(256), SMEM_SZ, stream>>>(
        occ, coord, wt, bq, bk, bv, bo, gamma, beta, out, attn_out);
}
